// CCELMHeadCE_19765439496674
// MI455X (gfx1250) — matrix operand order repaired and checked by reading, not compiled, not run on hardware
//
#include <hip/hip_runtime.h>
#include <hip/hip_bf16.h>

// Problem constants (match reference exactly; all tilings divide evenly)
#define HDIM 1024
#define VOCAB 50304
#define NTOK 4096
#define NSEG 8
#define SEGV (VOCAB / NSEG)           // 6288 vocab entries per segment
#define NSUB (SEGV / 16)              // 393 vocab subtiles of 16
#define SUBG 3                        // subtiles per A-fragment reuse group
#define NGRP (NSUB / SUBG)            // 131 groups (exact: 393 = 3*131)
#define TOKB 64                       // tokens per workgroup
#define NBLK (NTOK / TOKB)            // 64 workgroups
#define KSTEPS (HDIM / 32)            // 32 k-steps of 32 (bf16 WMMA K)
#define LDS_ROW_DW 516                // (1024 + 8 pad) bf16 per row -> dwords
#define IGNORE_INDEX (-100)

typedef __attribute__((ext_vector_type(16))) __bf16 v16bf;
typedef __attribute__((ext_vector_type(8)))  float  v8f;

union Frag {
    v16bf v;
    unsigned int u[8];
    uint4 q[2];
};

// One-time staging pack: round-half-up then v_perm_b32 (3 VALU ops).
__device__ __forceinline__ unsigned int pack_bf16_rn(float lo, float hi) {
    unsigned int ul = __float_as_uint(lo) + 0x8000u;
    unsigned int uh = __float_as_uint(hi) + 0x8000u;
    return __builtin_amdgcn_perm(uh, ul, 0x07060302u);
}

// Hot-loop pack (W stream): truncate-to-bf16, single v_perm_b32.
__device__ __forceinline__ unsigned int pack_bf16_tr(float lo, float hi) {
    return __builtin_amdgcn_perm(__float_as_uint(hi), __float_as_uint(lo),
                                 0x07060302u);
}

__global__ __launch_bounds__(256) void cce_main(
    const float* __restrict__ x, const float* __restrict__ W,
    float* __restrict__ ws_max, float* __restrict__ ws_sum) {
    __shared__ __align__(16) unsigned int lds_x[TOKB * LDS_ROW_DW]; // 132 KB bf16 x-tile

    const int tid  = threadIdx.x;
    const int lane = tid & 31;
    const int n    = lane & 15;   // output column within subtile
    const int hik  = lane >> 4;   // half-wave selector
    const int seg  = tid >> 5;    // wave id == vocab segment
    const int m0   = blockIdx.x * TOKB;

    // ---- Stage x tile (64 rows x 1024 f32, contiguous) into LDS as bf16 ----
    const float* xblk = x + (size_t)m0 * HDIM;
    for (int it = 0; it < 64; ++it) {
        int i = tid * 4 + it * 1024;                 // covers [0, 65536)
        float4 f = *reinterpret_cast<const float4*>(xblk + i);
        int row = i >> 10, off = i & 1023;
        unsigned int* p = &lds_x[row * LDS_ROW_DW + (off >> 1)];
        p[0] = pack_bf16_rn(f.x, f.y);
        p[1] = pack_bf16_rn(f.z, f.w);
    }
    __syncthreads();

    const int vb0 = seg * SEGV;

    // Running per-lane logsumexp state for the 32 (tile,row) pairs this lane
    // covers (rows r+8*hik of token subtiles t=0..3).
    float mres[32], sres[32];
#pragma unroll
    for (int i = 0; i < 32; ++i) { mres[i] = -1e30f; sres[i] = 0.f; }

    const v8f vzero = {0.f, 0.f, 0.f, 0.f, 0.f, 0.f, 0.f, 0.f};
    const float* wlane = W + (size_t)(vb0 + n) * HDIM + hik * 16;

    for (int grp = 0; grp < NGRP; ++grp) {
        v8f acc[SUBG * 4];
#pragma unroll
        for (int i = 0; i < SUBG * 4; ++i) acc[i] = vzero;

        const float* wrow0 = wlane + (size_t)((grp * SUBG + 0) * 16) * HDIM;
        const float* wrow1 = wlane + (size_t)((grp * SUBG + 1) * 16) * HDIM;
        const float* wrow2 = wlane + (size_t)((grp * SUBG + 2) * 16) * HDIM;

        for (int k = 0; k < KSTEPS; ++k) {
            // A fragments: loaded once per k-step, reused by SUBG=3 subtiles
            Frag a[4];
#pragma unroll
            for (int t = 0; t < 4; ++t) {
                const uint4* ap = reinterpret_cast<const uint4*>(
                    &lds_x[(t * 16 + n) * LDS_ROW_DW + k * 16 + hik * 8]);
                a[t].q[0] = ap[0];     // K = kb+hik*16 .. +7
                a[t].q[1] = ap[1];     // K = kb+hik*16+8 .. +15
            }
#pragma unroll
            for (int sg = 0; sg < SUBG; ++sg) {
                const float* wrow = (sg == 0) ? wrow0 : (sg == 1) ? wrow1 : wrow2;
                // B fragment: 16 consecutive f32 from this lane's W row -> bf16
                Frag b;
                const float4* wp = reinterpret_cast<const float4*>(wrow + k * 32);
                float4 f0 = wp[0], f1 = wp[1], f2 = wp[2], f3 = wp[3];
                b.u[0] = pack_bf16_tr(f0.x, f0.y); b.u[1] = pack_bf16_tr(f0.z, f0.w);
                b.u[2] = pack_bf16_tr(f1.x, f1.y); b.u[3] = pack_bf16_tr(f1.z, f1.w);
                b.u[4] = pack_bf16_tr(f2.x, f2.y); b.u[5] = pack_bf16_tr(f2.z, f2.w);
                b.u[6] = pack_bf16_tr(f3.x, f3.y); b.u[7] = pack_bf16_tr(f3.z, f3.w);
#pragma unroll
                for (int t = 0; t < 4; ++t) {
                    acc[sg * 4 + t] = __builtin_amdgcn_wmma_f32_16x16x32_bf16(
                        false, a[t].v, false, b.v, (short)0, acc[sg * 4 + t],
                        false, false);
                }
            }
        }
        // Online logsumexp update for this lane's columns (one exp per value)
#pragma unroll
        for (int sg = 0; sg < SUBG; ++sg)
#pragma unroll
            for (int t = 0; t < 4; ++t)
#pragma unroll
                for (int r = 0; r < 8; ++r) {
                    float v = acc[sg * 4 + t][r];
                    int i = t * 8 + r;
                    float d = v - mres[i];
                    float e = __expf(-fabsf(d));
                    sres[i] = (d > 0.f) ? (sres[i] * e + 1.f) : (sres[i] + e);
                    mres[i] = fmaxf(mres[i], v);
                }
    }

    // Merge (m,s) across the 16 lanes of each half-wave.
#pragma unroll
    for (int i = 0; i < 32; ++i) {
        float m = mres[i], s = sres[i];
#pragma unroll
        for (int msk = 1; msk < 16; msk <<= 1) {
            float mo = __shfl_xor(m, msk);
            float so = __shfl_xor(s, msk);
            float nm = fmaxf(m, mo);
            s = s * __expf(m - nm) + so * __expf(mo - nm);
            m = nm;
        }
        mres[i] = m; sres[i] = s;
    }

    if (n == 0) {  // lanes 0 and 16 write their rows
#pragma unroll
        for (int t = 0; t < 4; ++t)
#pragma unroll
            for (int r = 0; r < 8; ++r) {
                int row = m0 + t * 16 + r + hik * 8;
                int idx = row * NSEG + seg;
                int i = t * 8 + r;
                ws_max[idx] = mres[i];
                ws_sum[idx] = sres[i];
            }
    }
}

// Exact fp32 target logit: one wave per token computes dot(x[t], W[clip(y[t])]).
__global__ __launch_bounds__(256) void cce_target(
    const float* __restrict__ x, const float* __restrict__ W,
    const int* __restrict__ y, float* __restrict__ ws_tgt) {
    const int wave  = threadIdx.x >> 5;
    const int lane  = threadIdx.x & 31;
    const int token = blockIdx.x * 8 + wave;
    int yc = min(max(y[token], 0), VOCAB - 1);
    const float* xr = x + (size_t)token * HDIM;
    const float* wr = W + (size_t)yc * HDIM;
    float s = 0.f;
#pragma unroll
    for (int i = lane * 4; i < HDIM; i += 32 * 4) {
        float4 a = *reinterpret_cast<const float4*>(xr + i);
        float4 b = *reinterpret_cast<const float4*>(wr + i);
        s += a.x * b.x + a.y * b.y + a.z * b.z + a.w * b.w;
    }
#pragma unroll
    for (int msk = 1; msk < 32; msk <<= 1) s += __shfl_xor(s, msk);
    if (lane == 0) ws_tgt[token] = s;
}

__global__ __launch_bounds__(256) void cce_merge(
    const float* __restrict__ ws_max, const float* __restrict__ ws_sum,
    const float* __restrict__ ws_tgt, const int* __restrict__ y,
    float* __restrict__ part_loss, float* __restrict__ part_cnt) {
    __shared__ float sl[256];
    __shared__ float sc[256];
    int t = blockIdx.x * 256 + threadIdx.x;
    float gm = -1e30f;
    for (int j = 0; j < NSEG; ++j) gm = fmaxf(gm, ws_max[t * NSEG + j]);
    float S = 0.f;
    for (int j = 0; j < NSEG; ++j)
        S += ws_sum[t * NSEG + j] * __expf(ws_max[t * NSEG + j] - gm);
    float lse = gm + __logf(S);
    bool valid = (y[t] != IGNORE_INDEX);
    sl[threadIdx.x] = valid ? (lse - ws_tgt[t]) : 0.f;
    sc[threadIdx.x] = valid ? 1.f : 0.f;
    __syncthreads();
    for (int st = 128; st > 0; st >>= 1) {
        if (threadIdx.x < st) {
            sl[threadIdx.x] += sl[threadIdx.x + st];
            sc[threadIdx.x] += sc[threadIdx.x + st];
        }
        __syncthreads();
    }
    if (threadIdx.x == 0) {
        part_loss[blockIdx.x] = sl[0];
        part_cnt[blockIdx.x]  = sc[0];
    }
}

__global__ void cce_final(const float* __restrict__ pl,
                          const float* __restrict__ pc,
                          float* __restrict__ out, int nparts) {
    if (threadIdx.x == 0 && blockIdx.x == 0) {
        float L = 0.f, C = 0.f;
        for (int i = 0; i < nparts; ++i) { L += pl[i]; C += pc[i]; }
        out[0] = L / fmaxf(C, 1.f);
    }
}

extern "C" void kernel_launch(void* const* d_in, const int* in_sizes, int n_in,
                              void* d_out, int out_size, void* d_ws, size_t ws_size,
                              hipStream_t stream) {
    const float* x = (const float*)d_in[0];
    const float* W = (const float*)d_in[1];
    const int*   y = (const int*)d_in[2];
    float* out = (float*)d_out;

    // Workspace carve-out: 2 * N*NSEG + N + block partials (~279 KB)
    float* wsm = (float*)d_ws;
    float* wss = wsm + (size_t)NTOK * NSEG;
    float* wst = wss + (size_t)NTOK * NSEG;
    float* pl  = wst + (size_t)NTOK;
    float* pc  = pl + (NTOK / 256);

    cce_main<<<dim3(NBLK), dim3(256), 0, stream>>>(x, W, wsm, wss);
    cce_target<<<dim3(NTOK / 8), dim3(256), 0, stream>>>(x, W, y, wst);
    cce_merge<<<dim3(NTOK / 256), dim3(256), 0, stream>>>(wsm, wss, wst, y, pl, pc);
    cce_final<<<dim3(1), dim3(32), 0, stream>>>(pl, pc, out, NTOK / 256);
}
